// MMPrompt_inspired_23759759082002
// MI455X (gfx1250) — matrix operand order repaired and checked by reading, not compiled, not run on hardware
//
#include <hip/hip_runtime.h>
#include <hip/hip_bf16.h>

// ---------------------------------------------------------------------------
// MI455X (gfx1250) implementation.
// Dense GEMMs run v_wmma_f32_16x16x32_bf16 fed by double-buffered
// global_load_async_to_lds_b128 (ASYNCcnt-tracked CDNA5 async copies).
// All GEMM operands are pre-converted to bf16 (weights transposed to [N][K])
// so the hot loop contains zero conversion VALU work.
// Graph scatter/gather is f32 VALU + global atomics (bandwidth bound; the
// whole intermediate set is L2-resident at 192MB).
// Workspace requirement: ~376 MB.
// ---------------------------------------------------------------------------

#define N_ENTITY 30000
#define N_MOVIE  6000
#define DD       384
#define HH       768
#define E_KG     200000
#define E_C      200000
#define E_S      80000
#define NUM_REL  12
#define NUM_BASES 8
#define BB       16
#define LE       48
#define LT       256
#define NOUTCOL  18432   // N_LAYER*N_BLOCK*H

typedef __bf16 v16bf __attribute__((ext_vector_type(16)));
typedef float  v8f   __attribute__((ext_vector_type(8)));
typedef unsigned short ushort_t;

union ABFrag { unsigned int u[8]; v16bf v; };

static __device__ __forceinline__ unsigned short f2bf(float f) {
  unsigned int u = __float_as_uint(f);
  u += 0x7FFFu + ((u >> 16) & 1u);      // round-to-nearest-even
  return (unsigned short)(u >> 16);
}
static __device__ __forceinline__ unsigned int pack2bf(float lo, float hi) {
  return (unsigned int)f2bf(lo) | ((unsigned int)f2bf(hi) << 16);
}

// CDNA5 async copy: 16 bytes global -> LDS, tracked by ASYNCcnt.
static __device__ __forceinline__ void async_ld_b128(unsigned lds_addr,
                                                     unsigned voff,
                                                     const ushort_t* base) {
  asm volatile("global_load_async_to_lds_b128 %0, %1, %2"
               :: "v"(lds_addr), "v"(voff), "s"(base)
               : "memory");
}
static __device__ __forceinline__ unsigned lds_off(const void* p) {
  return (unsigned)(size_t)p;   // low 32 bits of flat LDS address = LDS offset
}

// ---------------------------------------------------------------------------
// WMMA bf16 GEMM: C = op(A[MxK] @ Bt[NxK]^T + bias (+res)).
// A, Bt are bf16 (Bt pre-transposed, K contiguous for both).
// Block: 256 threads = 8 waves; tile 128(M) x 64(N), BK=32, double buffered.
// ---------------------------------------------------------------------------
#define BM 128
#define BN 64
#define BK 32
#define LDS_STRIDE 40   // shorts per row (32 + 8 pad): 80B, 16B aligned

__global__ __launch_bounds__(256) void gemm_bf16_wmma(
    const ushort_t* __restrict__ A, const ushort_t* __restrict__ Bt,
    const float* __restrict__ bias, const float* __restrict__ Res,
    float* __restrict__ C, int M, int N, int K, int relu, int permuteStore)
{
  __shared__ ushort_t As[2][BM * LDS_STRIDE];
  __shared__ ushort_t Bs[2][BN * LDS_STRIDE];

  const int tid   = threadIdx.x;
  const int lane  = tid & 31;
  const int wave  = tid >> 5;
  const int waveM = wave & 3;     // 4 waves along M
  const int waveN = wave >> 2;    // 2 waves along N
  const int l16   = lane & 15;
  const int half  = lane >> 4;
  const int m0 = blockIdx.y * BM;
  const int n0 = blockIdx.x * BN;

  // staging coordinates: each thread moves one 16B chunk per async op
  const int aRow = tid >> 2;      // 0..63 (iter0), +64 (iter1)
  const int aSeg = tid & 3;       // 16B segment within the 64B K-chunk
  const int bRow = tid >> 2;      // 0..63
  const int bSeg = tid & 3;

  v8f acc[2][2];
#pragma unroll
  for (int mi = 0; mi < 2; ++mi)
#pragma unroll
    for (int ni = 0; ni < 2; ++ni)
#pragma unroll
      for (int i = 0; i < 8; ++i) acc[mi][ni][i] = 0.0f;

  const int nk = K / BK;

#define STAGE(ks, buf)                                                        \
  do {                                                                        \
    int k0_ = (ks) * BK;                                                      \
    int gm0_ = m0 + aRow;       if (gm0_ >= M) gm0_ = M - 1;                  \
    int gm1_ = m0 + aRow + 64;  if (gm1_ >= M) gm1_ = M - 1;                  \
    int gn_  = n0 + bRow;       if (gn_  >= N) gn_  = N - 1;                  \
    unsigned ka_ = (unsigned)(k0_ + aSeg * 8);                                \
    unsigned kb_ = (unsigned)(k0_ + bSeg * 8);                                \
    async_ld_b128(lds_off(&As[buf][aRow * LDS_STRIDE + aSeg * 8]),            \
                  ((unsigned)gm0_ * (unsigned)K + ka_) * 2u, A);              \
    async_ld_b128(lds_off(&As[buf][(aRow + 64) * LDS_STRIDE + aSeg * 8]),     \
                  ((unsigned)gm1_ * (unsigned)K + ka_) * 2u, A);              \
    async_ld_b128(lds_off(&Bs[buf][bRow * LDS_STRIDE + bSeg * 8]),            \
                  ((unsigned)gn_ * (unsigned)K + kb_) * 2u, Bt);              \
  } while (0)

  STAGE(0, 0);
  if (nk > 1) STAGE(1, 1);

  for (int ks = 0; ks < nk; ++ks) {
    const int cur = ks & 1;
    if (ks + 1 < nk) asm volatile("s_wait_asynccnt 3" ::: "memory");
    else             asm volatile("s_wait_asynccnt 0" ::: "memory");
    __syncthreads();                       // buf[cur] fully written (all waves)

    ABFrag afrag[2], bfrag[2];
#pragma unroll
    for (int mi = 0; mi < 2; ++mi) {
      int mrow = waveM * 32 + mi * 16 + l16;
#pragma unroll
      for (int v = 0; v < 8; ++v) {
        int kk = (v >> 2) * 16 + half * 8 + (v & 3) * 2;
        afrag[mi].u[v] = *(const unsigned int*)&As[cur][mrow * LDS_STRIDE + kk];
      }
    }
#pragma unroll
    for (int ni = 0; ni < 2; ++ni) {
      int ncol = waveN * 32 + ni * 16 + l16;
#pragma unroll
      for (int v = 0; v < 8; ++v) {
        int kk = (v >> 2) * 16 + half * 8 + (v & 3) * 2;
        bfrag[ni].u[v] = *(const unsigned int*)&Bs[cur][ncol * LDS_STRIDE + kk];
      }
    }
#pragma unroll
    for (int mi = 0; mi < 2; ++mi)
#pragma unroll
      for (int ni = 0; ni < 2; ++ni)
        acc[mi][ni] = __builtin_amdgcn_wmma_f32_16x16x32_bf16(
            false, afrag[mi].v, false, bfrag[ni].v,
            (short)0, acc[mi][ni], false, false);

    __syncthreads();                       // all waves done reading buf[cur]
    if (ks + 2 < nk) STAGE(ks + 2, cur);   // overwrite consumed buffer
  }
#undef STAGE

  // ---- epilogue: bias / relu / residual / (permuted NT) store ----
#pragma unroll
  for (int mi = 0; mi < 2; ++mi) {
#pragma unroll
    for (int ni = 0; ni < 2; ++ni) {
      int mbase = m0 + waveM * 32 + mi * 16 + half * 8;
      int n     = n0 + waveN * 32 + ni * 16 + l16;
      if (n >= N) continue;
#pragma unroll
      for (int i = 0; i < 8; ++i) {
        int m = mbase + i;
        if (m >= M) continue;
        float v = acc[mi][ni][i];
        if (bias) v += bias[n];
        if (relu) v = v > 0.f ? v : 0.f;
        if (Res)  v += Res[(size_t)m * N + n];
        if (!permuteStore) {
          C[(size_t)m * N + n] = v;
        } else {
          // prompt[b,t, l,blk,head,hd] -> out[l,blk,b,head,t,hd]; 302MB stream
          int b = m >> 8, t = m & 255;
          int lb = n / 768, rem = n - lb * 768;
          int head = rem >> 6, hd = rem & 63;
          size_t oidx = ((((size_t)lb * BB + b) * 12 + head) * LT + t) * 64 + hd;
          __builtin_nontemporal_store(v, &C[oidx]);
        }
      }
    }
  }
}

// ---------------------------------------------------------------------------
// Conversion kernels (bf16 operand preparation)
// ---------------------------------------------------------------------------
__global__ void k_cvt_bf16(const float* __restrict__ src, ushort_t* __restrict__ dst, int n) {
  int i = 4 * (blockIdx.x * blockDim.x + threadIdx.x);
  if (i + 3 < n) {
    float4 f = *(const float4*)(src + i);
    uint2 p; p.x = pack2bf(f.x, f.y); p.y = pack2bf(f.z, f.w);
    *(uint2*)(dst + i) = p;
  } else {
    for (int j = i; j < n; ++j) dst[j] = f2bf(src[j]);
  }
}
// src [K][N] f32 -> dst [N][K] bf16
__global__ void k_cvt_bf16_t(const float* __restrict__ src, ushort_t* __restrict__ dst,
                             int K, int N) {
  int id = blockIdx.x * blockDim.x + threadIdx.x;
  if (id >= K * N) return;
  int n = id / K, k = id - n * K;
  dst[id] = f2bf(src[(size_t)k * N + n]);
}
// WT[r][j][i] = bf16( sum_b comp[r,b] * bases[b][i][j] )  (transposed + bf16)
__global__ void k_compute_Wt(const float* __restrict__ bases,
                             const float* __restrict__ comp,
                             ushort_t* __restrict__ WT) {
  int idx = blockIdx.x * blockDim.x + threadIdx.x;
  const int per = DD * DD;
  if (idx >= NUM_REL * per) return;
  int r = idx / per, ji = idx - r * per;
  int j = ji / DD, i = ji - j * DD;
  float s = 0.f;
#pragma unroll
  for (int b = 0; b < NUM_BASES; ++b)
    s += comp[r * NUM_BASES + b] * bases[(size_t)b * per + i * DD + j];
  WT[idx] = f2bf(s);
}

// ---------------------------------------------------------------------------
// Elementwise / graph kernels
// ---------------------------------------------------------------------------
__global__ void k_fill(float* p, float v, int n) {
  int i = blockIdx.x * blockDim.x + threadIdx.x;
  if (i < n) p[i] = v;
}
__global__ void k_zero_int(int* p, int n) {
  int i = blockIdx.x * blockDim.x + threadIdx.x;
  if (i < n) p[i] = 0;
}
__global__ void k_count_rel(const int* __restrict__ dst, const int* __restrict__ et,
                            int* __restrict__ cnt, int E) {
  int e = blockIdx.x * blockDim.x + threadIdx.x;
  if (e < E) atomicAdd(&cnt[et[e] * N_ENTITY + dst[e]], 1);
}
__global__ void k_rgcn_scatter(const float* __restrict__ h, float* __restrict__ out,
                               const int* __restrict__ src, const int* __restrict__ dst,
                               const int* __restrict__ et, const int* __restrict__ cnt,
                               int r, int E) {
  int id = blockIdx.x * blockDim.x + threadIdx.x;
  if (id >= E * (DD / 4)) return;
  int e = id / (DD / 4);
  if (et[e] != r) return;
  int c = (id - e * (DD / 4)) * 4;
  int s = src[e], d = dst[e];
  float norm = 1.0f / fmaxf((float)cnt[r * N_ENTITY + d], 1.0f);
  const float4 xv = *(const float4*)(h + (size_t)s * DD + c);
  float* op = out + (size_t)d * DD + c;
  atomicAdd(op + 0, norm * xv.x);
  atomicAdd(op + 1, norm * xv.y);
  atomicAdd(op + 2, norm * xv.z);
  atomicAdd(op + 3, norm * xv.w);
}
__global__ void k_deg_edges(const int* __restrict__ dst, float* __restrict__ deg, int E) {
  int e = blockIdx.x * blockDim.x + threadIdx.x;
  if (e < E) atomicAdd(&deg[dst[e]], 1.0f);
}
__global__ void k_rsqrt(float* p, int n) {
  int i = blockIdx.x * blockDim.x + threadIdx.x;
  if (i < n) p[i] = rsqrtf(p[i]);
}
__global__ void k_gcn_init(const float* __restrict__ x, const float* __restrict__ dis,
                           float* __restrict__ out, int n) {
  int idx = blockIdx.x * blockDim.x + threadIdx.x;
  if (idx >= n * DD) return;
  int i = idx / DD;
  float d = dis[i];
  out[idx] = d * d * x[idx];
}
__global__ void k_gcn_scatter(const float* __restrict__ x, const float* __restrict__ dis,
                              float* __restrict__ out, const int* __restrict__ src,
                              const int* __restrict__ dst, int E) {
  int id = blockIdx.x * blockDim.x + threadIdx.x;
  if (id >= E * (DD / 4)) return;
  int e = id / (DD / 4);
  int c = (id - e * (DD / 4)) * 4;
  int s = src[e], d = dst[e];
  float w = dis[s] * dis[d];
  const float4 xv = *(const float4*)(x + (size_t)s * DD + c);
  float* op = out + (size_t)d * DD + c;
  atomicAdd(op + 0, w * xv.x);
  atomicAdd(op + 1, w * xv.y);
  atomicAdd(op + 2, w * xv.z);
  atomicAdd(op + 3, w * xv.w);
}
__global__ void k_gather_rows(const float* __restrict__ src, const int* __restrict__ idx,
                              float* __restrict__ dst, int n, int cols) {
  int id = blockIdx.x * blockDim.x + threadIdx.x;
  if (id >= n * cols) return;
  int i = id / cols, c = id - i * cols;
  dst[id] = src[(size_t)idx[i] * cols + c];
}
__global__ void k_addpair(float* __restrict__ o, const float* __restrict__ x,
                          const float* __restrict__ y, float s, int n, int accum) {
  int i = blockIdx.x * blockDim.x + threadIdx.x;
  if (i >= n) return;
  float v = s * (x[i] + y[i]);
  o[i] = accum ? (o[i] + v) : v;
}
__global__ void k_comb4(float* __restrict__ o, const float* __restrict__ a,
                        const float* __restrict__ b, const float* __restrict__ c,
                        const float* __restrict__ d, float s, int n) {
  int i = blockIdx.x * blockDim.x + threadIdx.x;
  if (i < n) o[i] = s * (a[i] + b[i] + c[i] + d[i]);
}
__global__ void k_scatter_add(const float* __restrict__ mmv, const int* __restrict__ m2e,
                              float* __restrict__ ent, int nm) {
  int id = blockIdx.x * blockDim.x + threadIdx.x;
  if (id >= nm * DD) return;
  int m = id / DD, c = id - m * DD;
  atomicAdd(&ent[(size_t)m2e[m] * DD + c], mmv[id]);
}

// one block per (b,t): 48-way cross-attention + softmax + fused residual
__global__ __launch_bounds__(128) void k_attention(
    const float* __restrict__ Q, const float* __restrict__ EE,
    const float* __restrict__ TOK, float* __restrict__ OUT) {
  __shared__ float qs[HH];
  __shared__ float as[LE];
  int bt = blockIdx.x;            // 0..4095
  int b = bt >> 8;
  int tid = threadIdx.x;
  const float* qrow = Q + (size_t)bt * HH;
  for (int c = tid; c < HH; c += 128) qs[c] = qrow[c];
  __syncthreads();
  if (tid < LE) {
    const float* er = EE + ((size_t)b * LE + tid) * HH;
    float s = 0.f;
    for (int k = 0; k < HH; ++k) s += qs[k] * er[k];
    as[tid] = s * (1.0f / (float)HH);
  }
  __syncthreads();
  if (tid == 0) {
    float mx = as[0];
    for (int e = 1; e < LE; ++e) mx = fmaxf(mx, as[e]);
    float sum = 0.f;
    for (int e = 0; e < LE; ++e) { float ev = __expf(as[e] - mx); as[e] = ev; sum += ev; }
    float inv = 1.0f / sum;
    for (int e = 0; e < LE; ++e) as[e] *= inv;
  }
  __syncthreads();
  for (int c = tid; c < HH; c += 128) {
    float acc = TOK[(size_t)bt * HH + c];
    for (int e = 0; e < LE; ++e) acc += as[e] * EE[((size_t)b * LE + e) * HH + c];
    OUT[(size_t)bt * HH + c] = acc;
  }
}

// ---------------------------------------------------------------------------
// Host launcher
// ---------------------------------------------------------------------------
static inline int cdiv(int a, int b) { return (a + b - 1) / b; }

static void gemm(const ushort_t* A, const ushort_t* Bt, const float* bias,
                 const float* Res, float* C, int M, int N, int K,
                 int relu, int perm, hipStream_t s) {
  dim3 g(cdiv(N, BN), cdiv(M, BM));
  gemm_bf16_wmma<<<g, 256, 0, s>>>(A, Bt, bias, Res, C, M, N, K, relu, perm);
}

extern "C" void kernel_launch(void* const* d_in, const int* in_sizes, int n_in,
                              void* d_out, int out_size, void* d_ws, size_t ws_size,
                              hipStream_t stream) {
  (void)in_sizes; (void)n_in; (void)out_size; (void)ws_size;
  const float* node_embeds = (const float*)d_in[0];
  const float* rgcn_bases  = (const float*)d_in[1];
  const float* rgcn_comp   = (const float*)d_in[2];
  const float* rgcn_root   = (const float*)d_in[3];
  const float* rgcn_bias   = (const float*)d_in[4];
  const float* ep1_w1 = (const float*)d_in[5];
  const float* ep1_b1 = (const float*)d_in[6];
  const float* ep1_w2 = (const float*)d_in[7];
  const float* ep1_b2 = (const float*)d_in[8];
  const float* ep2_w  = (const float*)d_in[9];
  const float* ep2_b  = (const float*)d_in[10];
  const float* tp1_w1 = (const float*)d_in[11];
  const float* tp1_b1 = (const float*)d_in[12];
  const float* tp1_w2 = (const float*)d_in[13];
  const float* tp1_b2 = (const float*)d_in[14];
  const float* tp2_w  = (const float*)d_in[15];
  const float* tp2_b  = (const float*)d_in[16];
  const float* ca_w   = (const float*)d_in[17];
  const float* pp1_w1 = (const float*)d_in[18];
  const float* pp1_b1 = (const float*)d_in[19];
  const float* pp1_w2 = (const float*)d_in[20];
  const float* pp1_b2 = (const float*)d_in[21];
  const float* pp2_w  = (const float*)d_in[22];
  const float* pp2_b  = (const float*)d_in[23];
  const float* token_embeds = (const float*)d_in[24];
  const int* ei_kg  = (const int*)d_in[25];
  const int* etype  = (const int*)d_in[26];
  const int* ei_c   = (const int*)d_in[27];
  const int* ei_ts  = (const int*)d_in[28];
  const int* ei_is  = (const int*)d_in[29];
  const int* m2e    = (const int*)d_in[30];
  const int* entity_ids = (const int*)d_in[31];
  float* out = (float*)d_out;

  const int* src_kg = ei_kg;          const int* dst_kg = ei_kg + E_KG;
  const int* src_c  = ei_c;           const int* dst_c  = ei_c  + E_C;
  const int* src_ts = ei_ts;          const int* dst_ts = ei_ts + E_S;
  const int* src_is = ei_is;          const int* dst_is = ei_is + E_S;

  // ---- f32 workspace region ----
  float* ws = (float*)d_ws;
  size_t o = 0;
  int*   CNT  = (int*)(ws + o); o += (size_t)NUM_REL * N_ENTITY;
  float* DISC = ws + o; o += N_ENTITY;
  float* DIST = ws + o; o += N_MOVIE;
  float* DISI = ws + o; o += N_MOVIE;
  float* NF   = ws + o; o += (size_t)N_MOVIE * DD;
  float* G1   = ws + o; o += (size_t)N_MOVIE * DD;
  float* G2   = ws + o; o += (size_t)N_MOVIE * DD;
  float* MM   = ws + o; o += (size_t)N_MOVIE * DD;
  float* B0   = ws + o; o += (size_t)N_ENTITY * DD;   // ent0
  float* B1   = ws + o; o += (size_t)N_ENTITY * DD;
  float* B2   = ws + o; o += (size_t)N_ENTITY * DD;
  float* B3   = ws + o; o += (size_t)N_ENTITY * DD;
  float* ENTH = ws + o; o += (size_t)N_ENTITY * HH;
  // token-phase buffers alias B0..B3 region (entity buffers dead by then)
  float* tb = B0;
  float* TH1    = tb; tb += (size_t)BB * LT * DD;
  float* TOKMID = tb; tb += (size_t)BB * LT * HH;
  float* TOK2   = tb; tb += (size_t)BB * LT * HH;
  float* Qb     = tb; tb += (size_t)BB * LT * HH;
  float* PR1    = tb; tb += (size_t)BB * LT * HH;
  float* PH     = tb; tb += (size_t)BB * LT * DD;
  float* PR2    = tb; tb += (size_t)BB * LT * HH;
  float* EEMB   = tb; tb += (size_t)BB * LE * HH;

  // ---- bf16 workspace region (transposed weights + activation arena) ----
  ushort_t* wb = (ushort_t*)(ws + o);
  ushort_t* rootT  = wb; wb += (size_t)DD * DD;
  ushort_t* WT     = wb; wb += (size_t)NUM_REL * DD * DD;
  ushort_t* ep1w1T = wb; wb += (size_t)DD * (DD / 2);
  ushort_t* ep1w2T = wb; wb += (size_t)(DD / 2) * DD;
  ushort_t* ep2T   = wb; wb += (size_t)DD * HH;
  ushort_t* tp1w1T = wb; wb += (size_t)HH * DD;
  ushort_t* tp1w2T = wb; wb += (size_t)DD * HH;
  ushort_t* tp2T   = wb; wb += (size_t)HH * HH;
  ushort_t* caT    = wb; wb += (size_t)HH * HH;
  ushort_t* pp1w1T = wb; wb += (size_t)HH * DD;
  ushort_t* pp1w2T = wb; wb += (size_t)DD * HH;
  ushort_t* pp2T   = wb; wb += (size_t)HH * NOUTCOL;
  ushort_t* Abf    = wb; wb += (size_t)N_ENTITY * DD;   // activation arena

  const int T = 256;
  const int EC4 = DD / 4;

#define CVT(srcp, dstp, nn) k_cvt_bf16<<<cdiv((nn), 4 * T), T, 0, stream>>>(srcp, dstp, nn)
#define CVTT(srcp, dstp, KK, NN) \
  k_cvt_bf16_t<<<cdiv((KK) * (NN), T), T, 0, stream>>>(srcp, dstp, KK, NN)

  // ---- one-time weight prep: transpose + bf16 ----
  k_compute_Wt<<<cdiv(NUM_REL * DD * DD, T), T, 0, stream>>>(rgcn_bases, rgcn_comp, WT);
  CVTT(rgcn_root, rootT, DD, DD);
  CVTT(ep1_w1, ep1w1T, DD, DD / 2);
  CVTT(ep1_w2, ep1w2T, DD / 2, DD);
  CVTT(ep2_w,  ep2T,   DD, HH);
  CVTT(tp1_w1, tp1w1T, HH, DD);
  CVTT(tp1_w2, tp1w2T, DD, HH);
  CVTT(tp2_w,  tp2T,   HH, HH);
  CVTT(ca_w,   caT,    HH, HH);
  CVTT(pp1_w1, pp1w1T, HH, DD);
  CVTT(pp1_w2, pp1w2T, DD, HH);
  CVTT(pp2_w,  pp2T,   HH, NOUTCOL);

  // ---- RGCN ----
  k_zero_int<<<cdiv(NUM_REL * N_ENTITY, T), T, 0, stream>>>(CNT, NUM_REL * N_ENTITY);
  k_count_rel<<<cdiv(E_KG, T), T, 0, stream>>>(dst_kg, etype, CNT, E_KG);
  CVT(node_embeds, Abf, N_ENTITY * DD);
  // ent0 = x @ root + bias + x  (residual folds the "+ node_embeds")
  gemm(Abf, rootT, rgcn_bias, node_embeds, B0, N_ENTITY, DD, DD, 0, 0, stream);
  for (int r = 0; r < NUM_REL; ++r) {
    gemm(Abf, WT + (size_t)r * DD * DD, nullptr, nullptr, B1, N_ENTITY, DD, DD, 0, 0, stream);
    k_rgcn_scatter<<<cdiv(E_KG * EC4, T), T, 0, stream>>>(B1, B0, src_kg, dst_kg, etype, CNT, r, E_KG);
  }

  // ---- movie feature GCNs ----
  k_gather_rows<<<cdiv(N_MOVIE * DD, T), T, 0, stream>>>(B0, m2e, NF, N_MOVIE, DD);
  k_fill<<<cdiv(N_MOVIE, T), T, 0, stream>>>(DIST, 1.0f, N_MOVIE);
  k_deg_edges<<<cdiv(E_S, T), T, 0, stream>>>(dst_ts, DIST, E_S);
  k_rsqrt<<<cdiv(N_MOVIE, T), T, 0, stream>>>(DIST, N_MOVIE);
  k_fill<<<cdiv(N_MOVIE, T), T, 0, stream>>>(DISI, 1.0f, N_MOVIE);
  k_deg_edges<<<cdiv(E_S, T), T, 0, stream>>>(dst_is, DISI, E_S);
  k_rsqrt<<<cdiv(N_MOVIE, T), T, 0, stream>>>(DISI, N_MOVIE);
  k_fill<<<cdiv(N_ENTITY, T), T, 0, stream>>>(DISC, 1.0f, N_ENTITY);
  k_deg_edges<<<cdiv(E_C, T), T, 0, stream>>>(dst_c, DISC, E_C);
  k_rsqrt<<<cdiv(N_ENTITY, T), T, 0, stream>>>(DISC, N_ENTITY);

  k_gcn_init<<<cdiv(N_MOVIE * DD, T), T, 0, stream>>>(NF, DIST, G1, N_MOVIE);
  k_gcn_scatter<<<cdiv(E_S * EC4, T), T, 0, stream>>>(NF, DIST, G1, src_ts, dst_ts, E_S);
  k_gcn_init<<<cdiv(N_MOVIE * DD, T), T, 0, stream>>>(G1, DIST, G2, N_MOVIE);
  k_gcn_scatter<<<cdiv(E_S * EC4, T), T, 0, stream>>>(G1, DIST, G2, src_ts, dst_ts, E_S);
  k_addpair<<<cdiv(N_MOVIE * DD, T), T, 0, stream>>>(MM, G1, G2, 0.25f, N_MOVIE * DD, 0);
  k_gcn_init<<<cdiv(N_MOVIE * DD, T), T, 0, stream>>>(NF, DISI, G1, N_MOVIE);
  k_gcn_scatter<<<cdiv(E_S * EC4, T), T, 0, stream>>>(NF, DISI, G1, src_is, dst_is, E_S);
  k_gcn_init<<<cdiv(N_MOVIE * DD, T), T, 0, stream>>>(G1, DISI, G2, N_MOVIE);
  k_gcn_scatter<<<cdiv(E_S * EC4, T), T, 0, stream>>>(G1, DISI, G2, src_is, dst_is, E_S);
  k_addpair<<<cdiv(N_MOVIE * DD, T), T, 0, stream>>>(MM, G1, G2, 0.25f, N_MOVIE * DD, 1);

  k_gcn_init<<<cdiv(N_ENTITY * DD, T), T, 0, stream>>>(B0, DISC, B1, N_ENTITY);
  k_gcn_scatter<<<cdiv(E_C * EC4, T), T, 0, stream>>>(B0, DISC, B1, src_c, dst_c, E_C);
  k_gcn_init<<<cdiv(N_ENTITY * DD, T), T, 0, stream>>>(B1, DISC, B2, N_ENTITY);
  k_gcn_scatter<<<cdiv(E_C * EC4, T), T, 0, stream>>>(B1, DISC, B2, src_c, dst_c, E_C);
  k_gcn_init<<<cdiv(N_ENTITY * DD, T), T, 0, stream>>>(B2, DISC, B3, N_ENTITY);
  k_gcn_scatter<<<cdiv(E_C * EC4, T), T, 0, stream>>>(B2, DISC, B3, src_c, dst_c, E_C);
  k_comb4<<<cdiv(N_ENTITY * DD, T), T, 0, stream>>>(B1, B1, B2, B3, B0, 0.25f, N_ENTITY * DD);
  k_scatter_add<<<cdiv(N_MOVIE * DD, T), T, 0, stream>>>(MM, m2e, B1, N_MOVIE);

  // ---- entity MLP + projection ----
  CVT(B1, Abf, N_ENTITY * DD);
  gemm(Abf, ep1w1T, ep1_b1, nullptr, B2, N_ENTITY, DD / 2, DD, 1, 0, stream);
  CVT(B2, Abf, N_ENTITY * (DD / 2));
  gemm(Abf, ep1w2T, ep1_b2, B1, B3, N_ENTITY, DD, DD / 2, 0, 0, stream);
  CVT(B3, Abf, N_ENTITY * DD);
  gemm(Abf, ep2T, ep2_b, nullptr, ENTH, N_ENTITY, HH, DD, 0, 0, stream);

  // ---- token path ----
  const int MT = BB * LT;   // 4096
  CVT(token_embeds, Abf, MT * HH);
  gemm(Abf, tp1w1T, tp1_b1, nullptr, TH1, MT, DD, HH, 1, 0, stream);
  CVT(TH1, Abf, MT * DD);
  gemm(Abf, tp1w2T, tp1_b2, token_embeds, TOKMID, MT, HH, DD, 0, 0, stream);
  CVT(TOKMID, Abf, MT * HH);
  gemm(Abf, tp2T, tp2_b, nullptr, TOK2, MT, HH, HH, 0, 0, stream);
  CVT(TOK2, Abf, MT * HH);
  gemm(Abf, caT, nullptr, nullptr, Qb, MT, HH, HH, 0, 0, stream);

  // ---- cross attention ----
  k_gather_rows<<<cdiv(BB * LE * HH, T), T, 0, stream>>>(ENTH, entity_ids, EEMB, BB * LE, HH);
  k_attention<<<MT, 128, 0, stream>>>(Qb, EEMB, TOK2, PR1);

  // ---- prompt MLP + final projection with fused transpose ----
  CVT(PR1, Abf, MT * HH);
  gemm(Abf, pp1w1T, pp1_b1, nullptr, PH, MT, DD, HH, 1, 0, stream);
  CVT(PH, Abf, MT * DD);
  gemm(Abf, pp1w2T, pp1_b2, PR1, PR2, MT, HH, DD, 0, 0, stream);
  CVT(PR2, Abf, MT * HH);
  gemm(Abf, pp2T, pp2_b, nullptr, out, MT, NOUTCOL, HH, 0, 1, stream);

#undef CVT
#undef CVTT
}
